// _RPN_21131239096448
// MI455X (gfx1250) — matrix operand order repaired and checked by reading, not compiled, not run on hardware
//
#include <hip/hip_runtime.h>
#include <math.h>

// ---------------- problem constants ----------------
#define BB     32
#define C512   512
#define TT     16
#define H7     7
#define W7     7
#define NPOS   49          // 7*7
#define AA     9
#define NANCH  441         // 49*9
#define POST_NMS 300
#define NMS_TH 0.7f

// output layout (floats): rois[32,300,5] | cls_prob[32,18,7,7] | bbox[32,36,7,7]
#define OUT_ROIS_OFF 0
#define OUT_CLS_OFF  48000
#define OUT_BBOX_OFF 76224

// ---------------- vector types for WMMA ----------------
typedef __attribute__((ext_vector_type(16))) __bf16        v16bf;
typedef __attribute__((ext_vector_type(8)))  float         v8f;
typedef __attribute__((ext_vector_type(4)))  unsigned int  v4u;

union Frag { v16bf h; v4u q[2]; };
union Acc  { v8f v; float f[8]; };

__device__ __forceinline__ unsigned short f2bf(float f) {
    unsigned int u = __float_as_uint(f);
    return (unsigned short)((u + 0x7FFFu + ((u >> 16) & 1u)) >> 16);  // RNE
}
__device__ __forceinline__ float bf2f(unsigned short h) {
    return __uint_as_float(((unsigned int)h) << 16);
}

// =====================================================================
// Prep kernels: fp32 -> bf16 with layout changes
// =====================================================================

// base_feat [B][C][T][7][7] f32  ->  in_tl [B][T][49][C] bf16 (channels-last)
__global__ void k_cvt_in(const float* __restrict__ src, unsigned short* __restrict__ dst, int n) {
    int i = blockIdx.x * blockDim.x + threadIdx.x;
    if (i >= n) return;
    int c = i % C512; int r = i / C512;
    int p = r % NPOS; r /= NPOS;
    int t = r % TT;   int b = r / TT;
    float v = src[(((size_t)b * C512 + c) * TT + t) * NPOS + p];
    dst[i] = f2bf(v);
}

// conv weights [O][C][jmax taps] f32 -> [O][jmax][C] bf16 (K = j*512 + c contiguous)
__global__ void k_cvt_w(const float* __restrict__ src, unsigned short* __restrict__ dst,
                        int jmax, int n) {
    int i = blockIdx.x * blockDim.x + threadIdx.x;
    if (i >= n) return;
    int c = i % C512; int r = i / C512;
    int j = r % jmax; int o = r / jmax;
    dst[i] = f2bf(src[((size_t)o * C512 + c) * jmax + j]);
}

// head weights: cls [18][512] ++ bbox [36][512] -> bf16 [54][512]
__global__ void k_cvt_wh(const float* __restrict__ cls_w, const float* __restrict__ bbox_w,
                         unsigned short* __restrict__ dst) {
    int i = blockIdx.x * blockDim.x + threadIdx.x;
    if (i >= 54 * C512) return;
    int c = i % C512; int o = i / C512;
    float v = (o < 18) ? cls_w[(size_t)o * C512 + c] : bbox_w[(size_t)(o - 18) * C512 + c];
    dst[i] = f2bf(v);
}

// device port of generate_anchors + shifts (441 anchors)
__global__ void k_anchors(float* __restrict__ anch) {
    int i = blockIdx.x * blockDim.x + threadIdx.x;
    if (i >= NANCH) return;
    int a = i % AA, k = i / AA;
    int ri = a / 3, si = a % 3;
    const float ratios[3] = {0.5f, 1.f, 2.f};
    const float scales[3] = {4.f, 8.f, 16.f};
    float ctr = 7.5f;
    float wsr = rintf(sqrtf(256.f / ratios[ri]));
    float hsr = rintf(wsr * ratios[ri]);
    float x1 = ctr - 0.5f * (wsr - 1.f), y1 = ctr - 0.5f * (hsr - 1.f);
    float x2 = ctr + 0.5f * (wsr - 1.f), y2 = ctr + 0.5f * (hsr - 1.f);
    float w  = x2 - x1 + 1.f, h = y2 - y1 + 1.f;
    float xc = x1 + 0.5f * (w - 1.f), yc = y1 + 0.5f * (h - 1.f);
    float W2 = w * scales[si], H2 = h * scales[si];
    int gy = k / 7, gx = k % 7;
    float sxf = gx * 16.f, syf = gy * 16.f;
    anch[i * 4 + 0] = xc - 0.5f * (W2 - 1.f) + sxf;
    anch[i * 4 + 1] = yc - 0.5f * (H2 - 1.f) + syf;
    anch[i * 4 + 2] = xc + 0.5f * (W2 - 1.f) + sxf;
    anch[i * 4 + 3] = yc + 0.5f * (H2 - 1.f) + syf;
}

// =====================================================================
// Implicit-GEMM conv as bf16 WMMA.
// CONV==1: rpn_conv 3x3x3, pad(1,1,1): grid (B*T) blocks;   K = 27*512.
//          epilogue: bias+ReLU+bf16 store (channels-last).
// CONV==2: t16 16x3x3, pad(0,1,1):     grid (B, KSPLIT=8);  K = 144*512.
//          each block owns a K-range; fp32 partial tile stored to its own
//          slice (deterministic split-K), reduced by k_fin2.
// Block = 256 threads = 8 waves. M = 49->64 rows, wave w owns N cols [64w,64w+64).
// A tile (64x32 bf16) double-buffered in LDS; B streamed from global/L2.
// =====================================================================
template <int CONV>
__global__ __launch_bounds__(256) void conv_wmma(
    const unsigned short* __restrict__ inA,   // channels-last bf16 activations
    const unsigned short* __restrict__ wB,    // [O][K] bf16 weights
    const float* __restrict__ bias,           // [512] f32 (CONV==1 only)
    unsigned short* __restrict__ outA,        // channels-last bf16 output (CONV==1)
    float* __restrict__ partial)              // [KSPLIT][B*49*512] f32 (CONV==2)
{
    constexpr int KTOT   = (CONV == 1) ? 27 * 512 : 144 * 512;
    constexpr int KSTEPS = KTOT / 32;
    constexpr int KSPLIT = (CONV == 1) ? 1 : 8;
    constexpr int KCHUNK = KSTEPS / KSPLIT;
    __shared__ unsigned short lA[2][64 * 40];   // stride 40 bf16 (80B) breaks bank conflicts

    const int tid = threadIdx.x;
    int b, tz, ky;
    if (CONV == 1) { b = blockIdx.x >> 4; tz = blockIdx.x & 15; ky = 0; }
    else           { b = blockIdx.x;      tz = 0;               ky = blockIdx.y; }
    const int s_begin = ky * KCHUNK;
    const int s_end   = s_begin + KCHUNK;

    // ---- staging role: thread t loads one 16B chunk of the 64x32 A tile
    const int srow = tid >> 2;          // 0..63 (M row)
    const int ssub = tid & 3;           // 16B chunk within 32 bf16
    const int sy = srow / W7, sx = srow % W7;
    const bool rvalid = srow < NPOS;

    auto stage_load = [&](int s) -> v4u {
        v4u r = {0u, 0u, 0u, 0u};
        int j  = s >> 4;                 // tap index (512/32 = 16 steps per tap)
        int c0 = (s & 15) << 5;          // channel base
        int dz = j / 9, rem = j % 9, dy = rem / 3, dx = rem % 3;
        int iy = sy + dy - 1, ix = sx + dx - 1;
        int iz = (CONV == 1) ? (tz + dz - 1) : dz;
        bool v = rvalid && (iy >= 0) && (iy < H7) && (ix >= 0) && (ix < W7);
        if (CONV == 1) v = v && (iz >= 0) && (iz < TT);
        if (v) {
            size_t off = ((size_t)((b * TT + iz) * NPOS + iy * W7 + ix)) * C512 + c0 + ssub * 8;
            r = *(const v4u*)(inA + off);
        }
        return r;
    };

    // ---- compute role
    const int wave  = tid >> 5;
    const int lane  = tid & 31;
    const int lrow  = lane & 15;
    const int khalf = lane >> 4;        // which 16-K half this lane holds
    const int nbase = wave * 64;

    Acc acc[4][4];
#pragma unroll
    for (int i = 0; i < 4; ++i)
#pragma unroll
        for (int j2 = 0; j2 < 4; ++j2)
            acc[i][j2].v = (v8f){0.f, 0.f, 0.f, 0.f, 0.f, 0.f, 0.f, 0.f};

    v4u pend = stage_load(s_begin);
    int buf = 0;
    for (int s = s_begin; s < s_end; ++s) {
        *(v4u*)&lA[buf][srow * 40 + ssub * 8] = pend;
        __syncthreads();
        if (s + 1 < s_end) pend = stage_load(s + 1);   // overlap global load w/ compute

        const int k0 = s * 32;
        if (s + 4 < s_end)   // -> global_prefetch_b8 for the weight stream
            __builtin_prefetch(wB + (size_t)(nbase + lrow) * KTOT + (size_t)(s + 4) * 32, 0, 0);

        // B fragments: lane holds K chunks [kb,kb+8) and [kb+16,kb+24) of one column (channels-contiguous weights)
        Frag bf[4];
#pragma unroll
        for (int nt = 0; nt < 4; ++nt) {
            size_t boff = (size_t)(nbase + nt * 16 + lrow) * KTOT + k0 + khalf * 8;
            bf[nt].q[0] = *(const v4u*)(wB + boff);
            bf[nt].q[1] = *(const v4u*)(wB + boff + 16);
        }
        // A fragments from LDS: lane row = lrow, K chunks [kb,kb+8) and [kb+16,kb+24)
#pragma unroll
        for (int mt = 0; mt < 4; ++mt) {
            Frag af;
            int aoff = (mt * 16 + lrow) * 40 + khalf * 8;
            af.q[0] = *(const v4u*)&lA[buf][aoff];
            af.q[1] = *(const v4u*)&lA[buf][aoff + 16];
#pragma unroll
            for (int nt = 0; nt < 4; ++nt) {
                acc[mt][nt].v = __builtin_amdgcn_wmma_f32_16x16x32_bf16(
                    false, af.h, false, bf[nt].h, (short)0, acc[mt][nt].v, false, false);
            }
        }
        __syncthreads();
        buf ^= 1;
    }

    // ---- epilogue
    if (CONV == 1) {
        // bias + ReLU + bf16, channels-last store
#pragma unroll
        for (int mt = 0; mt < 4; ++mt) {
#pragma unroll
            for (int nt = 0; nt < 4; ++nt) {
                int o = nbase + nt * 16 + lrow;
                float bo = bias[o];
#pragma unroll
                for (int v = 0; v < 8; ++v) {
                    int m = mt * 16 + v + (khalf ? 8 : 0);
                    if (m < NPOS) {
                        float val = fmaxf(acc[mt][nt].f[v] + bo, 0.f);
                        size_t pos = (size_t)(b * TT + tz) * NPOS + m;
                        outA[pos * C512 + o] = f2bf(val);
                    }
                }
            }
        }
    } else {
        // deterministic split-K: store fp32 partial into this K-range's own slice
        float* pp = partial + (size_t)ky * ((size_t)BB * NPOS * C512);
#pragma unroll
        for (int mt = 0; mt < 4; ++mt) {
#pragma unroll
            for (int nt = 0; nt < 4; ++nt) {
                int o = nbase + nt * 16 + lrow;
#pragma unroll
                for (int v = 0; v < 8; ++v) {
                    int m = mt * 16 + v + (khalf ? 8 : 0);
                    if (m < NPOS) {
                        size_t pos = (size_t)b * NPOS + m;
                        pp[pos * C512 + o] = acc[mt][nt].f[v];
                    }
                }
            }
        }
    }
}

// reduce the 8 split-K partials in fixed order, add bias, ReLU, convert to bf16
__global__ void k_fin2(const float* __restrict__ partial, const float* __restrict__ bias,
                       unsigned short* __restrict__ f16, int n) {
    int i = blockIdx.x * blockDim.x + threadIdx.x;
    if (i >= n) return;
    int o = i % C512;
    const size_t stride = (size_t)BB * NPOS * C512;
    float s = 0.f;
#pragma unroll
    for (int k = 0; k < 8; ++k) s += partial[(size_t)k * stride + i];
    f16[i] = f2bf(fmaxf(s + bias[o], 0.f));
}

// =====================================================================
// 1x1 heads: 54 output channels (18 cls + 36 bbox), K = 512 dot products.
// =====================================================================
__global__ void k_head(const unsigned short* __restrict__ f16,
                       const unsigned short* __restrict__ wh,
                       const float* __restrict__ cls_b,
                       const float* __restrict__ bbox_b,
                       float* __restrict__ cls_score,    // ws [B][49][18]
                       float* __restrict__ out_bbox)     // d_out bbox region [B][36][49]
{
    int i = blockIdx.x * blockDim.x + threadIdx.x;
    if (i >= BB * NPOS * 54) return;
    int o = i % 54; int r = i / 54; int m = r % NPOS; int b = r / NPOS;
    const unsigned short* fp = f16 + (size_t)(b * NPOS + m) * C512;
    const unsigned short* wp = wh + (size_t)o * C512;
    float acc = 0.f;
    for (int c = 0; c < C512; ++c) acc = fmaf(bf2f(fp[c]), bf2f(wp[c]), acc);
    if (o < 18) cls_score[(size_t)(b * NPOS + m) * 18 + o] = acc + cls_b[o];
    else        out_bbox[((size_t)b * 36 + (o - 18)) * NPOS + m] = acc + bbox_b[o - 18];
}

// pairwise (fg/bg) softmax over channel pairs (a, a+9)
__global__ void k_softmax(const float* __restrict__ cls_score, float* __restrict__ out_cls) {
    int i = blockIdx.x * blockDim.x + threadIdx.x;
    if (i >= BB * NPOS * AA) return;
    int a = i % AA; int r = i / AA; int m = r % NPOS; int b = r / NPOS;
    float s0 = cls_score[(size_t)(b * NPOS + m) * 18 + a];
    float s1 = cls_score[(size_t)(b * NPOS + m) * 18 + 9 + a];
    float mx = fmaxf(s0, s1);
    float e0 = expf(s0 - mx), e1 = expf(s1 - mx);
    float inv = 1.f / (e0 + e1);
    out_cls[((size_t)b * 18 + a) * NPOS + m]     = e0 * inv;
    out_cls[((size_t)b * 18 + 9 + a) * NPOS + m] = e1 * inv;
}

// =====================================================================
// Proposal layer: decode + clip + rank-sort + greedy NMS, one block per batch.
// =====================================================================
__global__ __launch_bounds__(256) void k_proposal(
    const float* __restrict__ out_cls, const float* __restrict__ out_bbox,
    const float* __restrict__ anch, const float* __restrict__ im_info,
    float* __restrict__ rois)
{
    __shared__ float bx[NANCH][4];
    __shared__ float sc[NANCH];
    __shared__ float area[NANCH];
    __shared__ short sorted[NANCH];
    __shared__ unsigned char keep[NANCH];

    int b = blockIdx.x, tid = threadIdx.x;
    float imh = im_info[b * 3 + 0], imw = im_info[b * 3 + 1];

    for (int i = tid; i < NANCH; i += blockDim.x) {
        int a = i % AA, m = i / AA;
        float s  = out_cls[((size_t)b * 18 + 9 + a) * NPOS + m];   // fg score
        float d0 = out_bbox[((size_t)b * 36 + a * 4 + 0) * NPOS + m];
        float d1 = out_bbox[((size_t)b * 36 + a * 4 + 1) * NPOS + m];
        float d2 = out_bbox[((size_t)b * 36 + a * 4 + 2) * NPOS + m];
        float d3 = out_bbox[((size_t)b * 36 + a * 4 + 3) * NPOS + m];
        float ax1 = anch[i * 4 + 0], ay1 = anch[i * 4 + 1];
        float ax2 = anch[i * 4 + 2], ay2 = anch[i * 4 + 3];
        float aw = ax2 - ax1 + 1.f, ah = ay2 - ay1 + 1.f;
        float acx = ax1 + 0.5f * aw, acy = ay1 + 0.5f * ah;
        float pcx = d0 * aw + acx, pcy = d1 * ah + acy;
        float pw = expf(d2) * aw, ph = expf(d3) * ah;
        float x1 = fminf(fmaxf(pcx - 0.5f * pw, 0.f), imw - 1.f);
        float y1 = fminf(fmaxf(pcy - 0.5f * ph, 0.f), imh - 1.f);
        float x2 = fminf(fmaxf(pcx + 0.5f * pw, 0.f), imw - 1.f);
        float y2 = fminf(fmaxf(pcy + 0.5f * ph, 0.f), imh - 1.f);
        bx[i][0] = x1; bx[i][1] = y1; bx[i][2] = x2; bx[i][3] = y2;
        sc[i] = s;
        area[i] = (x2 - x1 + 1.f) * (y2 - y1 + 1.f);
        keep[i] = 1;
    }
    __syncthreads();

    // stable descending rank sort (argsort(-scores))
    for (int i = tid; i < NANCH; i += blockDim.x) {
        float si = sc[i]; int rk = 0;
        for (int j = 0; j < NANCH; ++j) {
            float sj = sc[j];
            if (sj > si || (sj == si && j < i)) ++rk;
        }
        sorted[rk] = (short)i;
    }
    __syncthreads();

    // greedy NMS over sorted order; inner suppression is wave-parallel
    for (int ii = 0; ii < NANCH - 1; ++ii) {
        if (keep[ii]) {
            int bi = sorted[ii];
            float bx1 = bx[bi][0], by1 = bx[bi][1], bx2 = bx[bi][2], by2 = bx[bi][3];
            float ba = area[bi];
            for (int jj = ii + 1 + tid; jj < NANCH; jj += blockDim.x) {
                if (keep[jj]) {
                    int bj = sorted[jj];
                    float xx1 = fmaxf(bx1, bx[bj][0]);
                    float yy1 = fmaxf(by1, bx[bj][1]);
                    float xx2 = fminf(bx2, bx[bj][2]);
                    float yy2 = fminf(by2, bx[bj][3]);
                    float iw = fmaxf(xx2 - xx1 + 1.f, 0.f);
                    float ih = fmaxf(yy2 - yy1 + 1.f, 0.f);
                    float inter = iw * ih;
                    float iou = inter / (ba + area[bj] - inter);
                    if (iou > NMS_TH) keep[jj] = 0;
                }
            }
        }
        __syncthreads();
    }

    if (tid == 0) {
        float* rb = rois + (size_t)b * POST_NMS * 5;
        int cnt = 0;
        for (int ii = 0; ii < NANCH && cnt < POST_NMS; ++ii) {
            if (keep[ii]) {
                int bi = sorted[ii];
                rb[cnt * 5 + 0] = (float)b;
                rb[cnt * 5 + 1] = bx[bi][0]; rb[cnt * 5 + 2] = bx[bi][1];
                rb[cnt * 5 + 3] = bx[bi][2]; rb[cnt * 5 + 4] = bx[bi][3];
                ++cnt;
            }
        }
        for (; cnt < POST_NMS; ++cnt) {
            rb[cnt * 5 + 0] = (float)b;   // bidx column kept even for padding (matches ref)
            rb[cnt * 5 + 1] = 0.f; rb[cnt * 5 + 2] = 0.f;
            rb[cnt * 5 + 3] = 0.f; rb[cnt * 5 + 4] = 0.f;
        }
    }
}

// =====================================================================
// host launcher
// =====================================================================
extern "C" void kernel_launch(void* const* d_in, const int* in_sizes, int n_in,
                              void* d_out, int out_size, void* d_ws, size_t ws_size,
                              hipStream_t stream)
{
    const float* base_feat = (const float*)d_in[0];
    const float* im_info   = (const float*)d_in[1];
    const float* rpn_w     = (const float*)d_in[4];
    const float* rpn_b     = (const float*)d_in[5];
    const float* t16_w     = (const float*)d_in[6];
    const float* t16_b     = (const float*)d_in[7];
    const float* cls_w     = (const float*)d_in[8];
    const float* cls_b     = (const float*)d_in[9];
    const float* bbox_w    = (const float*)d_in[10];
    const float* bbox_b    = (const float*)d_in[11];

    // workspace layout (bytes)
    const size_t N_IN   = (size_t)BB * TT * NPOS * C512;            // 12,845,056 bf16
    const size_t N_W1   = (size_t)C512 * 27 * C512;                 //  7,077,888 bf16
    const size_t N_W2   = (size_t)C512 * 144 * C512;                // 37,748,736 bf16
    const size_t N_F16  = (size_t)BB * NPOS * C512;                 //    802,816 bf16
    const size_t OFF_IN   = 0;
    const size_t OFF_W1   = OFF_IN   + N_IN  * 2;
    const size_t OFF_OUT1 = OFF_W1   + N_W1  * 2;
    const size_t OFF_W2   = OFF_OUT1 + N_IN  * 2;
    const size_t OFF_F16  = OFF_W2   + N_W2  * 2;
    const size_t OFF_WH   = OFF_F16  + N_F16 * 2;
    const size_t OFF_CS   = OFF_WH   + (size_t)54 * C512 * 2 + 128;
    const size_t OFF_ANCH = OFF_CS   + (size_t)BB * NPOS * 18 * 4;
    const size_t OFF_P2   = OFF_ANCH + (size_t)NANCH * 4 * 4 + 128;  // [8][B*49*512] f32

    char* ws = (char*)d_ws;
    unsigned short* in_tl  = (unsigned short*)(ws + OFF_IN);
    unsigned short* w1r    = (unsigned short*)(ws + OFF_W1);
    unsigned short* out1   = (unsigned short*)(ws + OFF_OUT1);
    unsigned short* w2r    = (unsigned short*)(ws + OFF_W2);
    unsigned short* f16    = (unsigned short*)(ws + OFF_F16);
    unsigned short* wh     = (unsigned short*)(ws + OFF_WH);
    float*          cscore = (float*)(ws + OFF_CS);
    float*          anch   = (float*)(ws + OFF_ANCH);
    float*          part2  = (float*)(ws + OFF_P2);

    float* out      = (float*)d_out;
    float* out_rois = out + OUT_ROIS_OFF;
    float* out_cls  = out + OUT_CLS_OFF;
    float* out_bbox = out + OUT_BBOX_OFF;

    // 1) precision / layout prep
    {
        int n = (int)N_IN;
        k_cvt_in<<<(n + 255) / 256, 256, 0, stream>>>(base_feat, in_tl, n);
    }
    {
        int n = (int)N_W1;
        k_cvt_w<<<(n + 255) / 256, 256, 0, stream>>>(rpn_w, w1r, 27, n);
    }
    {
        int n = (int)N_W2;
        k_cvt_w<<<(n + 255) / 256, 256, 0, stream>>>(t16_w, w2r, 144, n);
    }
    k_cvt_wh<<<(54 * C512 + 255) / 256, 256, 0, stream>>>(cls_w, bbox_w, wh);
    k_anchors<<<2, 256, 0, stream>>>(anch);

    // 2) the two big convolutions as bf16 WMMA implicit GEMMs
    conv_wmma<1><<<BB * TT, 256, 0, stream>>>(in_tl, w1r, rpn_b, out1, nullptr);
    conv_wmma<2><<<dim3(BB, 8), 256, 0, stream>>>(out1, w2r, nullptr, nullptr, part2);
    {
        int n = (int)N_F16;
        k_fin2<<<(n + 255) / 256, 256, 0, stream>>>(part2, t16_b, f16, n);
    }

    // 3) 1x1 heads, softmax, proposals
    {
        int n = BB * NPOS * 54;
        k_head<<<(n + 255) / 256, 256, 0, stream>>>(f16, wh, cls_b, bbox_b, cscore, out_bbox);
    }
    {
        int n = BB * NPOS * AA;
        k_softmax<<<(n + 255) / 256, 256, 0, stream>>>(cscore, out_cls);
    }
    k_proposal<<<BB, 256, 0, stream>>>(out_cls, out_bbox, anch, im_info, out_rois);

    (void)in_sizes; (void)n_in; (void)out_size; (void)ws_size;
}
